// BTDG_31774168055963
// MI455X (gfx1250) — compile-verified, matrix operand order checked by reading, not run on hardware
//
#include <hip/hip_runtime.h>
#include <hip/hip_bf16.h>
#include <cstdint>
#include <cstddef>

// ---------------------------------------------------------------------------
// Problem constants (from reference): B=2048, E=20000, R2=500, T=365, C=12,
// ed=rd=200.  Flat Tucker K = rd*ed = 40000 (exactly 1250 steps of 32).
// ---------------------------------------------------------------------------
#define ED      200
#define RD      200
#define BATCH   2048
#define EVOC    20000
#define NC      12
#define KFLAT   (RD * ED)          // 40000
#define KSLICES 10
#define SLICE_K (KFLAT / KSLICES)  // 4000 -> 125 k-steps of 32 per slice
#define KPAD    416                // 400 padded to 13 * 32
#define OCROWS  20096              // 157 * 128 (padded N rows of concat output emb)

typedef __attribute__((ext_vector_type(16))) __bf16 v16bf;
typedef __attribute__((ext_vector_type(8)))  float  v8f;

// ---------------------------------------------------------------------------
// CDNA5 async global->LDS copy (raw 16B per lane) + async counter wait.
// ---------------------------------------------------------------------------
__device__ __forceinline__ void async_copy16(unsigned lds_off, const void* gptr) {
    unsigned long long ga = (unsigned long long)(size_t)gptr;
    asm volatile("global_load_async_to_lds_b128 %0, %1, off"
                 :: "v"(lds_off), "v"(ga) : "memory");
}
__device__ __forceinline__ void wait_async_le(int n) {
    if (n == 4) asm volatile("s_wait_asynccnt 0x4" ::: "memory");
    else        asm volatile("s_wait_asynccnt 0x0" ::: "memory");
}

__device__ __forceinline__ unsigned lds_addr32(const void* p) {
    // flat shared pointer: low 32 bits are the LDS byte offset
    return (unsigned)(size_t)p;
}

// ---------------------------------------------------------------------------
// BatchNorm column statistics: one block (256 thr) per column.
//   val(b) = X[(gidx? gidx[b] : b)*200 + col] * (mul ? mul[midx[b]*200+col] : 1)
// ---------------------------------------------------------------------------
__global__ __launch_bounds__(256) void stats_kernel(
    const float* __restrict__ X, const int* __restrict__ gidx,
    const float* __restrict__ mul, const int* __restrict__ midx,
    float* __restrict__ meanOut, float* __restrict__ rstdOut)
{
    const int c = blockIdx.x, tid = threadIdx.x;
    float s = 0.f, ss = 0.f;
    for (int b = tid; b < BATCH; b += 256) {
        int row = gidx ? gidx[b] : b;
        float v = X[(size_t)row * ED + c];
        if (mul) v *= mul[(size_t)midx[b] * ED + c];
        s += v; ss += v * v;
    }
    __shared__ float rs[256], rq[256];
    rs[tid] = s; rq[tid] = ss; __syncthreads();
    for (int o = 128; o > 0; o >>= 1) {
        if (tid < o) { rs[tid] += rs[tid + o]; rq[tid] += rq[tid + o]; }
        __syncthreads();
    }
    if (tid == 0) {
        float m = rs[0] * (1.f / BATCH);
        float var = rq[0] * (1.f / BATCH) - m * m;
        meanOut[c] = m;
        rstdOut[c] = rsqrtf(var + 1e-5f);
    }
}

// ---------------------------------------------------------------------------
// cb[b] = fine2coarse[times[b]]
// ---------------------------------------------------------------------------
__global__ __launch_bounds__(256) void coarse_kernel(
    const int* __restrict__ times, const int* __restrict__ f2c, int* __restrict__ cb)
{
    int b = blockIdx.x * 256 + threadIdx.x;
    if (b < BATCH) cb[b] = f2c[times[b]];
}

// ---------------------------------------------------------------------------
// Deterministic stable group-by-coarse-id (C=12, B=2048: trivially serial).
// ---------------------------------------------------------------------------
__global__ void group_kernel(const int* __restrict__ cb,
                             int* __restrict__ offs, int* __restrict__ order)
{
    if (threadIdx.x == 0 && blockIdx.x == 0) {
        int counts[NC], cur[NC];
        for (int c = 0; c < NC; ++c) counts[c] = 0;
        for (int b = 0; b < BATCH; ++b) counts[cb[b]]++;
        int acc = 0;
        for (int c = 0; c < NC; ++c) { offs[c] = acc; cur[c] = acc; acc += counts[c]; }
        offs[NC] = acc;
        for (int b = 0; b < BATCH; ++b) { int c = cb[b]; order[cur[c]++] = b; }
    }
}

// ---------------------------------------------------------------------------
// Build bf16 activation rows:
//  A1 = bn(S1[s]) ; A2 = bn(S2[s]) * T_S[times] ; PP1 = P1[p] ; PP2 = P2[p]
// ---------------------------------------------------------------------------
__global__ __launch_bounds__(256) void prep_kernel(
    const int* __restrict__ s, const int* __restrict__ p, const int* __restrict__ times,
    const float* __restrict__ S1, const float* __restrict__ S2,
    const float* __restrict__ P1, const float* __restrict__ P2,
    const float* __restrict__ T_S,
    const float* __restrict__ g11, const float* __restrict__ b11,
    const float* __restrict__ g21, const float* __restrict__ b21,
    const float* __restrict__ mE1, const float* __restrict__ rE1,
    const float* __restrict__ mE2, const float* __restrict__ rE2,
    __bf16* __restrict__ A1, __bf16* __restrict__ A2,
    __bf16* __restrict__ PP1, __bf16* __restrict__ PP2)
{
    int idx = blockIdx.x * 256 + threadIdx.x;       // < BATCH*ED
    int b = idx / ED, j = idx - b * ED;
    int sb = s[b], pb = p[b], tb = times[b];
    float x1 = (S1[(size_t)sb * ED + j] - mE1[j]) * rE1[j] * g11[j] + b11[j];
    float x2 = (S2[(size_t)sb * ED + j] - mE2[j]) * rE2[j] * g21[j] + b21[j];
    A1[idx]  = (__bf16)x1;
    A2[idx]  = (__bf16)(x2 * T_S[(size_t)tb * ED + j]);
    PP1[idx] = (__bf16)P1[(size_t)pb * RD + j];
    PP2[idx] = (__bf16)P2[(size_t)pb * RD + j];
}

// ---------------------------------------------------------------------------
// Tucker-contraction GEMM:
//   out[b,f] += sum_k PP[b,k/200]*A[b,k%200] * G[k,f]     (K split KSLICES ways)
// A built on the fly; B (=G slab) staged via float4 global loads into
// registers (software pipeline), then stored pre-swizzled into the exact
// v_wmma_f32_16x16x32_bf16 fragment layout (ISA 7.12.2). LDS double-buffered.
// Block tile: M=64, N=224 (pad of 200); 8 waves = 4 m-subtiles x 2 n-halves.
// Grouped mode (order/offsets non-null): rows = order[...] within group grp,
// G slab = G2[grp].
// ---------------------------------------------------------------------------
__global__ __launch_bounds__(256) void tucker_kernel(
    const __bf16* __restrict__ Amat, const __bf16* __restrict__ PP,
    const float* __restrict__ G, float* __restrict__ outPart,
    const int* __restrict__ order, const int* __restrict__ offsets)
{
    const int tile = blockIdx.x & 31;
    const int grp  = blockIdx.x >> 5;
    int gstart, gend;
    const float* Gm;
    if (offsets) {
        gstart = offsets[grp]; gend = offsets[grp + 1];
        Gm = G + (size_t)grp * (size_t)KFLAT * ED;
    } else {
        gstart = 0; gend = BATCH; Gm = G;
    }
    const int tileStart = gstart + tile * 64;
    if (tileStart >= gend) return;                       // uniform early exit

    float* outSlice = outPart + (size_t)blockIdx.y * (BATCH * ED);

    // fragment-layout staging buffers: [buf][subtile][lane][16 elems]
    __shared__ __attribute__((aligned(32))) __bf16 As2[2][4][32][16];    //  8 KB
    __shared__ __attribute__((aligned(32))) __bf16 Bs2[2][14][32][16];   // 28 KB

    const int tid = threadIdx.x, lane = tid & 31, w = tid >> 5;
    const int msub = w & 3, nhalf = w >> 2;

    v8f acc[7] = {};

    // per-thread staging registers
    float4 breg[7];          // B tile: 32 x 224 / 4-wide = 1792 chunks / 256 thr
    float  areg[8];          // A tile: 64 x 32 = 2048 elems / 256 thr

    // global -> registers for k-step at k0
    auto loadAB = [&](int k0) {
#pragma unroll
        for (int i = 0; i < 7; ++i) {
            int idx = tid + i * 256;                 // chunk over 32 rows x 56 float4
            int kk = idx / 56, c4 = idx - kk * 56;
            int nn = c4 << 2;
            float4 v = {0.f, 0.f, 0.f, 0.f};
            if (nn < ED)                              // nn <= 196 -> nn+3 < 200
                v = *(const float4*)&Gm[(size_t)(k0 + kk) * ED + nn];
            breg[i] = v;
        }
#pragma unroll
        for (int i = 0; i < 8; ++i) {
            int idx = tid + i * 256;                 // 64 x 32
            int mm = idx >> 5, kk = idx & 31;
            int gr = tileStart + mm;
            float v = 0.f;
            if (gr < gend) {
                int b = order ? order[gr] : gr;
                int k = k0 + kk;
                int r = k / ED, e = k - r * ED;
                v = (float)PP[(size_t)b * RD + r] * (float)Amat[(size_t)b * ED + e];
            }
            areg[i] = v;
        }
    };
    // registers -> LDS (swizzled into WMMA fragment layout)
    auto storeAB = [&](int buf) {
#pragma unroll
        for (int i = 0; i < 7; ++i) {
            int idx = tid + i * 256;
            int kk = idx / 56, c4 = idx - kk * 56;
            int nn = c4 << 2;
            int t  = nn >> 4;                         // same group for all 4 comps
            int L0 = (nn & 15) + ((kk & 8) ? 16 : 0);
            int j  = (kk & 7) + ((kk & 16) ? 8 : 0);
            Bs2[buf][t][L0 + 0][j] = (__bf16)breg[i].x;
            Bs2[buf][t][L0 + 1][j] = (__bf16)breg[i].y;
            Bs2[buf][t][L0 + 2][j] = (__bf16)breg[i].z;
            Bs2[buf][t][L0 + 3][j] = (__bf16)breg[i].w;
        }
#pragma unroll
        for (int i = 0; i < 8; ++i) {
            int idx = tid + i * 256;
            int mm = idx >> 5, kk = idx & 31;
            int sub = mm >> 4;
            int L = (mm & 15) + ((kk & 8) ? 16 : 0);
            int j = (kk & 7) + ((kk & 16) ? 8 : 0);
            As2[buf][sub][L][j] = (__bf16)areg[i];
        }
    };

    const int kbase = blockIdx.y * SLICE_K;
    loadAB(kbase);
    storeAB(0);
    for (int step = 0; step < SLICE_K / 32; ++step) {
        const int buf = step & 1;
        if (step + 1 < SLICE_K / 32) loadAB(kbase + (step + 1) * 32); // overlap
        __syncthreads();                              // buf stores visible
        // ---- compute: 1 A-fragment, 7 B-fragments, 7 WMMAs
        v16bf af = *(const v16bf*)&As2[buf][msub][lane][0];
#pragma unroll
        for (int t = 0; t < 7; ++t) {
            v16bf bfv = *(const v16bf*)&Bs2[buf][nhalf * 7 + t][lane][0];
            acc[t] = __builtin_amdgcn_wmma_f32_16x16x32_bf16(
                false, af, false, bfv, (short)0, acc[t], false, false);
        }
        __syncthreads();                              // reads done before overwrite
        if (step + 1 < SLICE_K / 32) storeAB(1 - buf);
    }

    // ---- epilogue: plain stores into this k-slice's partial buffer
    const int lm = lane & 15, lk = lane >> 4;
#pragma unroll
    for (int t = 0; t < 7; ++t) {
#pragma unroll
        for (int v = 0; v < 8; ++v) {
            int n = nhalf * 112 + t * 16 + lm;
            int gr = tileStart + msub * 16 + lk * 8 + v;
            if (n < ED && gr < gend) {
                int b = order ? order[gr] : gr;
                outSlice[(size_t)b * ED + n] = acc[t][v];
            }
        }
    }
}

// ---------------------------------------------------------------------------
// Sum the KSLICES partial buffers (deterministic, no fp atomics).
// ---------------------------------------------------------------------------
__global__ __launch_bounds__(256) void reduce_kernel(
    const float* __restrict__ part1, const float* __restrict__ part2,
    float* __restrict__ pe1, float* __restrict__ pe2)
{
    int i = blockIdx.x * 256 + threadIdx.x;              // < BATCH*ED
    float a = 0.f, b = 0.f;
#pragma unroll
    for (int sl = 0; sl < KSLICES; ++sl) {
        a += part1[(size_t)sl * (BATCH * ED) + i];
        b += part2[(size_t)sl * (BATCH * ED) + i];
    }
    pe1[i] = a; pe2[i] = b;
}

// ---------------------------------------------------------------------------
// PE[b, 0:200]   = bn(pe1)        * g12 + b12
// PE[b, 200:400] = bn(pe2 * T_O)  * g22 + b22 ; PE[b,400:416] = 0     (bf16)
// ---------------------------------------------------------------------------
__global__ __launch_bounds__(256) void build_pe_kernel(
    const float* __restrict__ pe1, const float* __restrict__ pe2,
    const float* __restrict__ T_O, const int* __restrict__ times,
    const float* __restrict__ mP1, const float* __restrict__ rP1,
    const float* __restrict__ mP2, const float* __restrict__ rP2,
    const float* __restrict__ g12, const float* __restrict__ b12,
    const float* __restrict__ g22, const float* __restrict__ b22,
    __bf16* __restrict__ PE)
{
    int idx = blockIdx.x * 256 + threadIdx.x;            // < BATCH*KPAD
    int b = idx / KPAD, j = idx - b * KPAD;
    float v = 0.f;
    if (j < ED) {
        v = (pe1[(size_t)b * ED + j] - mP1[j]) * rP1[j] * g12[j] + b12[j];
    } else if (j < 2 * ED) {
        int jj = j - ED;
        float raw = pe2[(size_t)b * ED + jj] * T_O[(size_t)times[b] * ED + jj];
        v = (raw - mP2[jj]) * rP2[jj] * g22[jj] + b22[jj];
    }
    PE[idx] = (__bf16)v;
}

// ---------------------------------------------------------------------------
// OC[e, 0:200] = O1[e]; OC[e, 200:400] = O2[e]; zero pad (cols & rows) (bf16)
// ---------------------------------------------------------------------------
__global__ __launch_bounds__(256) void build_oc_kernel(
    const float* __restrict__ O1, const float* __restrict__ O2, __bf16* __restrict__ OC)
{
    int idx = blockIdx.x * 256 + threadIdx.x;            // < OCROWS*KPAD
    int e = idx / KPAD, j = idx - e * KPAD;
    float v = 0.f;
    if (e < EVOC) {
        if (j < ED)          v = O1[(size_t)e * ED + j];
        else if (j < 2 * ED) v = O2[(size_t)e * ED + (j - ED)];
    }
    OC[idx] = (__bf16)v;
}

// ---------------------------------------------------------------------------
// Final GEMM + sigmoid: out[b,e] = sigmoid( PE[b,:] . OC[e,:] )
// M=2048, N=20000(pad 20096), K=416. Block tile 128x128, 8 waves (4m x 2n),
// each wave 32x64 = 2x4 WMMA tiles. Double-buffered async global->LDS staging
// directly into fragment layout (data already bf16: raw 16B chunks line up).
// ---------------------------------------------------------------------------
__global__ __launch_bounds__(256) void final_gemm_kernel(
    const __bf16* __restrict__ PE, const __bf16* __restrict__ OC,
    float* __restrict__ out)
{
    __shared__ __attribute__((aligned(32))) __bf16 As2[2][8][32][16];  // 16 KB
    __shared__ __attribute__((aligned(32))) __bf16 Bs2[2][8][32][16];  // 16 KB

    const int tid = threadIdx.x, lane = tid & 31, w = tid >> 5;
    const int nbase = blockIdx.x * 128, mbase = blockIdx.y * 128;
    const int wm = w & 3, wn = w >> 2;

    v8f acc[2][4] = {};

    auto issue = [&](int buf, int k0) {
#pragma unroll
        for (int i = 0; i < 2; ++i) {
            int id  = tid + i * 256;                     // 512 16B chunks per matrix
            int sub = id >> 6, rem = id & 63, L = rem >> 1, jh = rem & 1;
            int kk  = ((L >> 4) << 3) + (jh << 4);       // fragment k-base of chunk
            const __bf16* srcA = PE + (size_t)(mbase + sub * 16 + (L & 15)) * KPAD + k0 + kk;
            async_copy16(lds_addr32(&As2[buf][sub][L][jh << 3]), srcA);
            const __bf16* srcB = OC + (size_t)(nbase + sub * 16 + (L & 15)) * KPAD + k0 + kk;
            async_copy16(lds_addr32(&Bs2[buf][sub][L][jh << 3]), srcB);
        }
    };

    issue(0, 0);
    for (int st = 0; st < KPAD / 32; ++st) {
        const int buf = st & 1;
        if (st + 1 < KPAD / 32) { issue(1 - buf, (st + 1) * 32); wait_async_le(4); }
        else                    {                                 wait_async_le(0); }
        __syncthreads();

        v16bf af[2], bfv[4];
#pragma unroll
        for (int im = 0; im < 2; ++im)
            af[im] = *(const v16bf*)&As2[buf][wm * 2 + im][lane][0];
#pragma unroll
        for (int in = 0; in < 4; ++in)
            bfv[in] = *(const v16bf*)&Bs2[buf][wn * 4 + in][lane][0];
#pragma unroll
        for (int im = 0; im < 2; ++im)
#pragma unroll
            for (int in = 0; in < 4; ++in)
                acc[im][in] = __builtin_amdgcn_wmma_f32_16x16x32_bf16(
                    false, af[im], false, bfv[in], (short)0, acc[im][in], false, false);
        __syncthreads();
    }

    const int lm = lane & 15, lk = lane >> 4;
#pragma unroll
    for (int im = 0; im < 2; ++im)
#pragma unroll
        for (int in = 0; in < 4; ++in)
#pragma unroll
            for (int v = 0; v < 8; ++v) {
                int m = mbase + wm * 32 + im * 16 + lk * 8 + v;
                int n = nbase + wn * 64 + in * 16 + lm;
                if (n < EVOC) {
                    float x = acc[im][in][v];
                    out[(size_t)m * EVOC + n] = 1.f / (1.f + __expf(-x));
                }
            }
}

// ---------------------------------------------------------------------------
// Host-side orchestration.
// Input order: s, p, o, times, fine2coarse, S1, O1, S2, O2, P1, P2, G1, G2,
//              T_S, T_O, g11, b11, g12, b12, g21, b21, g22, b22
// ---------------------------------------------------------------------------
extern "C" void kernel_launch(void* const* d_in, const int* in_sizes, int n_in,
                              void* d_out, int out_size, void* d_ws, size_t ws_size,
                              hipStream_t stream)
{
    const int*   s     = (const int*)  d_in[0];
    const int*   p     = (const int*)  d_in[1];
    const int*   times = (const int*)  d_in[3];
    const int*   f2c   = (const int*)  d_in[4];
    const float* S1    = (const float*)d_in[5];
    const float* O1    = (const float*)d_in[6];
    const float* S2    = (const float*)d_in[7];
    const float* O2    = (const float*)d_in[8];
    const float* P1    = (const float*)d_in[9];
    const float* P2    = (const float*)d_in[10];
    const float* G1    = (const float*)d_in[11];
    const float* G2    = (const float*)d_in[12];
    const float* T_S   = (const float*)d_in[13];
    const float* T_O   = (const float*)d_in[14];
    const float* g11   = (const float*)d_in[15];
    const float* b11   = (const float*)d_in[16];
    const float* g12   = (const float*)d_in[17];
    const float* b12   = (const float*)d_in[18];
    const float* g21   = (const float*)d_in[19];
    const float* b21   = (const float*)d_in[20];
    const float* g22   = (const float*)d_in[21];
    const float* b22   = (const float*)d_in[22];

    // ---- workspace carve-up (~58 MB), 256B aligned blocks
    char* w = (char*)d_ws;
    size_t off = 0;
    auto take = [&](size_t bytes) {
        char* q = w + off;
        off = (off + bytes + 255) & ~(size_t)255;
        return (void*)q;
    };
    float* mE1 = (float*)take(ED * 4);
    float* rE1 = (float*)take(ED * 4);
    float* mE2 = (float*)take(ED * 4);
    float* rE2 = (float*)take(ED * 4);
    float* mP1 = (float*)take(ED * 4);
    float* rP1 = (float*)take(ED * 4);
    float* mP2 = (float*)take(ED * 4);
    float* rP2 = (float*)take(ED * 4);
    int*   cb    = (int*)take(BATCH * 4);
    int*   offs  = (int*)take(16 * 4);
    int*   order = (int*)take(BATCH * 4);
    __bf16* A1  = (__bf16*)take((size_t)BATCH * ED * 2);
    __bf16* A2  = (__bf16*)take((size_t)BATCH * ED * 2);
    __bf16* PP1 = (__bf16*)take((size_t)BATCH * RD * 2);
    __bf16* PP2 = (__bf16*)take((size_t)BATCH * RD * 2);
    float* part1 = (float*)take((size_t)KSLICES * BATCH * ED * 4);
    float* part2 = (float*)take((size_t)KSLICES * BATCH * ED * 4);
    float* pe1   = (float*)take((size_t)BATCH * ED * 4);
    float* pe2   = (float*)take((size_t)BATCH * ED * 4);
    __bf16* PE = (__bf16*)take((size_t)BATCH * KPAD * 2);
    __bf16* OC = (__bf16*)take((size_t)OCROWS * KPAD * 2);
    (void)ws_size; (void)n_in; (void)in_sizes; (void)out_size;

    // 1) BN stats of gathered embeddings
    stats_kernel<<<ED, 256, 0, stream>>>(S1, s, nullptr, nullptr, mE1, rE1);
    stats_kernel<<<ED, 256, 0, stream>>>(S2, s, nullptr, nullptr, mE2, rE2);

    // 2) coarse ids + deterministic grouping
    coarse_kernel<<<(BATCH + 255) / 256, 256, 0, stream>>>(times, f2c, cb);
    group_kernel<<<1, 32, 0, stream>>>(cb, offs, order);

    // 3) bf16 activation rows
    prep_kernel<<<BATCH * ED / 256, 256, 0, stream>>>(
        s, p, times, S1, S2, P1, P2, T_S, g11, b11, g21, b21,
        mE1, rE1, mE2, rE2, A1, A2, PP1, PP2);

    // 4) Tucker GEMMs (branch1 ungrouped, branch2 grouped by coarse id)
    tucker_kernel<<<dim3(32, KSLICES), 256, 0, stream>>>(
        A1, PP1, G1, part1, nullptr, nullptr);
    tucker_kernel<<<dim3(NC * 32, KSLICES), 256, 0, stream>>>(
        A2, PP2, G2, part2, order, offs);

    // 5) reduce K-slice partials
    reduce_kernel<<<BATCH * ED / 256, 256, 0, stream>>>(part1, part2, pe1, pe2);

    // 6) BN stats of pe1 and pe2*T_O
    stats_kernel<<<ED, 256, 0, stream>>>(pe1, nullptr, nullptr, nullptr, mP1, rP1);
    stats_kernel<<<ED, 256, 0, stream>>>(pe2, nullptr, T_O, times, mP2, rP2);

    // 7) build concat operands
    build_pe_kernel<<<BATCH * KPAD / 256, 256, 0, stream>>>(
        pe1, pe2, T_O, times, mP1, rP1, mP2, rP2, g12, b12, g22, b22, PE);
    build_oc_kernel<<<OCROWS * KPAD / 256, 256, 0, stream>>>(O1, O2, OC);

    // 8) fused output GEMM + sigmoid
    final_gemm_kernel<<<dim3(OCROWS / 128, BATCH / 128), 256, 0, stream>>>(
        PE, OC, (float*)d_out);
}
